// GAT_44504451121630
// MI455X (gfx1250) — compile-verified, matrix operand order checked by reading, not run on hardware
//
#include <hip/hip_runtime.h>
#include <hip/hip_bf16.h>

typedef _Float16 h16;
typedef __attribute__((ext_vector_type(16))) _Float16 v16h;
typedef __attribute__((ext_vector_type(8)))  float    v8f;

#define NN 50000
#define EE 800000
#define QQ 200000
#define CC 128
#define NEG_SLOPE 0.2f
#define BN_EPS 1e-5f

// ---- order-preserving float<->uint encoding for atomicMax-based segment max ----
__device__ __forceinline__ unsigned enc_f32(float f) {
  unsigned u = __float_as_uint(f);
  return (u & 0x80000000u) ? ~u : (u | 0x80000000u);
}
__device__ __forceinline__ float dec_f32(unsigned u) {
  unsigned v = (u & 0x80000000u) ? (u & 0x7FFFFFFFu) : ~u;
  return __uint_as_float(v);
}

// ---- repack 128x128 f32 weight into f16 WMMA B-fragment order ----
// out[((nt*4 + kc)*32 + lane)*16 + i]  =  W[K*128 + nt*16 + (lane&15)]
//   lane<16 : i<8 -> K = kc*32 + i        ; i>=8 -> K = kc*32 + 16 + (i-8)
//   lane>=16: i<8 -> K = kc*32 + 8 + i    ; i>=8 -> K = kc*32 + 24 + (i-8)
__global__ void pack_w16_kernel(const float* __restrict__ W, h16* __restrict__ Wpk) {
  int o = blockIdx.x * blockDim.x + threadIdx.x;
  if (o >= CC * CC) return;
  int i  = o & 15;
  int l  = (o >> 4) & 31;
  int kc = (o >> 9) & 3;
  int nt = o >> 11;
  int lm = l & 15;
  int kh = (l >> 4) * 8;
  int K  = kc * 32 + kh + ((i < 8) ? i : (16 + (i - 8)));
  Wpk[o] = (h16)W[K * CC + nt * 16 + lm];
}

// ---- zero the scatter accumulator + per-node softmax state ----
__global__ void init_layer_kernel(float* __restrict__ acc, float* __restrict__ den,
                                  unsigned* __restrict__ emax, int nelem) {
  int i = blockIdx.x * blockDim.x + threadIdx.x;
  if (i < nelem) acc[i] = 0.0f;
  if (i < NN) { den[i] = 0.0f; emax[i] = 0x007FFFFFu; /* enc(-inf) */ }
}

// ---- H = X @ W  (rows x 128 @ 128x128), WMMA f16 in / f32 out ----
// block: 256 threads = 8 waves; block handles a 16-row M-tile; wave w covers N-tile w*16.
// B comes pre-packed in fragment order: one aligned 32B load per K-chunk.
__global__ __launch_bounds__(256) void gemm_node_f16(
    const float* __restrict__ X, const h16* __restrict__ Wpk, float* __restrict__ H) {
  __shared__ h16 As[16 * CC];
  int row0 = blockIdx.x * 16;
  #pragma unroll
  for (int j = 0; j < 8; ++j) {
    int idx = threadIdx.x + 256 * j;
    int r = idx >> 7, c = idx & 127;
    As[idx] = (h16)X[(row0 + r) * CC + c];
  }
  __syncthreads();
  int l  = threadIdx.x & 31;
  int wv = threadIdx.x >> 5;
  int lm = l & 15;
  int kh = (l >> 4) * 8;
  int ntile = wv * 16;
  const v16h* Bfrag = (const v16h*)Wpk;
  v8f acc = {};
  #pragma unroll
  for (int kc = 0; kc < 4; ++kc) {
    v16h a;
    #pragma unroll
    for (int i = 0; i < 8; ++i) {
      a[i]     = As[lm * CC + kc * 32 + kh + i];
      a[i + 8] = As[lm * CC + kc * 32 + 16 + kh + i];
    }
    v16h b = Bfrag[(wv * 4 + kc) * 32 + l];
    acc = __builtin_amdgcn_wmma_f32_16x16x32_f16(false, a, false, b, (short)0, acc, false, false);
  }
  #pragma unroll
  for (int r = 0; r < 8; ++r) {
    int m = r + ((l >> 4) ? 8 : 0);
    H[(row0 + m) * CC + ntile + lm] = acc[r];
  }
}

// ---- per-node attention scores: s_src = h . a_src, s_dst = h . a_dst (1 wave / node) ----
__global__ __launch_bounds__(256) void node_scores_kernel(
    const float* __restrict__ H, const float* __restrict__ asrc, const float* __restrict__ adst,
    float* __restrict__ ssrc, float* __restrict__ sdst) {
  int node = blockIdx.x * 8 + (threadIdx.x >> 5);
  if (node >= NN) return;
  int l = threadIdx.x & 31;
  float s1 = 0.0f, s2 = 0.0f;
  #pragma unroll
  for (int j = 0; j < 4; ++j) {
    float h = H[node * CC + l + 32 * j];
    s1 += h * asrc[l + 32 * j];
    s2 += h * adst[l + 32 * j];
  }
  #pragma unroll
  for (int off = 16; off > 0; off >>= 1) {
    s1 += __shfl_down(s1, off, 32);
    s2 += __shfl_down(s2, off, 32);
  }
  if (l == 0) { ssrc[node] = s1; sdst[node] = s2; }
}

__device__ __forceinline__ float edge_score(const float* ssrc, const float* sdst, int s, int d) {
  float e = ssrc[s] + sdst[d];
  return e > 0.0f ? e : NEG_SLOPE * e;
}

// ---- pass A: segment max over dst ----
__global__ void edge_max_kernel(const int* __restrict__ src, const int* __restrict__ dst,
                                const float* __restrict__ ssrc, const float* __restrict__ sdst,
                                unsigned* __restrict__ emax) {
  int i = blockIdx.x * blockDim.x + threadIdx.x;
  if (i >= EE) return;
  int d = dst[i];
  atomicMax(&emax[d], enc_f32(edge_score(ssrc, sdst, src[i], d)));
}

// ---- pass B: softmax denominator ----
__global__ void edge_den_kernel(const int* __restrict__ src, const int* __restrict__ dst,
                                const float* __restrict__ ssrc, const float* __restrict__ sdst,
                                const unsigned* __restrict__ emax, float* __restrict__ den) {
  int i = blockIdx.x * blockDim.x + threadIdx.x;
  if (i >= EE) return;
  int d = dst[i];
  float w = __expf(edge_score(ssrc, sdst, src[i], d) - dec_f32(emax[d]));
  atomicAdd(&den[d], w);
}

// ---- pass C: acc[dst] += alpha * h[src], one wave per edge, 4 channels per lane ----
__global__ __launch_bounds__(256) void edge_agg_kernel(
    const int* __restrict__ src, const int* __restrict__ dst,
    const float* __restrict__ ssrc, const float* __restrict__ sdst,
    const unsigned* __restrict__ emax, const float* __restrict__ den,
    const float* __restrict__ H, float* __restrict__ acc) {
  int e = blockIdx.x * 8 + (threadIdx.x >> 5);
  if (e >= EE) return;
  int l = threadIdx.x & 31;
  int s = src[e], d = dst[e];
  // overlap the 512B gathered row fetch with the softmax scalar math
  __builtin_prefetch(&H[(size_t)s * CC + l * 4], 0, 0);
  float w = __expf(edge_score(ssrc, sdst, s, d) - dec_f32(emax[d]));
  float alpha = w / (den[d] + 1e-16f);
  #pragma unroll
  for (int j = 0; j < 4; ++j) {
    int c = l + 32 * j;
    atomicAdd(&acc[d * CC + c], alpha * H[s * CC + c]);
  }
}

// ---- conv bias + batchnorm (eval) + relu, in place ----
__global__ void bn_relu_kernel(float* __restrict__ x, const float* __restrict__ b1,
                               const float* __restrict__ gamma, const float* __restrict__ beta,
                               const float* __restrict__ rmean, const float* __restrict__ rvar) {
  int i = blockIdx.x * blockDim.x + threadIdx.x;
  if (i >= NN * CC) return;
  int c = i & 127;
  float v = x[i] + b1[c];
  v = (v - rmean[c]) * rsqrtf(rvar[c] + BN_EPS) * gamma[c] + beta[c];
  x[i] = v > 0.0f ? v : 0.0f;
}

// ---- fused link predictor: p=(x2[e0]+b2)*(x2[e1]+b2); relu(p@Wp1+bp1); sigmoid(.@Wp2+bp2)
// block: 256 threads = 8 waves, 16 queries; WMMA 16x128 @ 128x128 with packed B.
__global__ __launch_bounds__(256) void link_pred_kernel(
    const float* __restrict__ x2, const float* __restrict__ b2,
    const h16* __restrict__ Wp1pk, const float* __restrict__ bp1,
    const float* __restrict__ Wp2, const float* __restrict__ bp2,
    const int* __restrict__ e0, const int* __restrict__ e1, float* __restrict__ out) {
  __shared__ h16 pA[16 * CC];
  __shared__ float hS[16 * CC];
  int q0 = blockIdx.x * 16;
  #pragma unroll
  for (int j = 0; j < 8; ++j) {
    int idx = threadIdx.x + 256 * j;
    int q = idx >> 7, c = idx & 127;
    int na = e0[q0 + q], nb = e1[q0 + q];
    float va = x2[na * CC + c] + b2[c];
    float vb = x2[nb * CC + c] + b2[c];
    pA[idx] = (h16)(va * vb);
  }
  __syncthreads();
  int l  = threadIdx.x & 31;
  int wv = threadIdx.x >> 5;
  int lm = l & 15;
  int kh = (l >> 4) * 8;
  int ntile = wv * 16;
  const v16h* Bfrag = (const v16h*)Wp1pk;
  v8f acc = {};
  #pragma unroll
  for (int kc = 0; kc < 4; ++kc) {
    v16h a;
    #pragma unroll
    for (int i = 0; i < 8; ++i) {
      a[i]     = pA[lm * CC + kc * 32 + kh + i];
      a[i + 8] = pA[lm * CC + kc * 32 + 16 + kh + i];
    }
    v16h b = Bfrag[(wv * 4 + kc) * 32 + l];
    acc = __builtin_amdgcn_wmma_f32_16x16x32_f16(false, a, false, b, (short)0, acc, false, false);
  }
  #pragma unroll
  for (int r = 0; r < 8; ++r) {
    int m = r + ((l >> 4) ? 8 : 0);
    float v = acc[r] + bp1[ntile + lm];
    hS[m * CC + ntile + lm] = v > 0.0f ? v : 0.0f;
  }
  __syncthreads();
  if (threadIdx.x < 16) {
    float s = bp2[0];
    #pragma unroll 8
    for (int j = 0; j < CC; ++j) s += hS[threadIdx.x * CC + j] * Wp2[j];
    out[q0 + threadIdx.x] = 1.0f / (1.0f + __expf(-s));
  }
}

extern "C" void kernel_launch(void* const* d_in, const int* in_sizes, int n_in,
                              void* d_out, int out_size, void* d_ws, size_t ws_size,
                              hipStream_t stream) {
  const float* emb    = (const float*)d_in[0];
  const float* W1     = (const float*)d_in[1];
  const float* a_src1 = (const float*)d_in[2];
  const float* a_dst1 = (const float*)d_in[3];
  const float* b1     = (const float*)d_in[4];
  const float* gamma  = (const float*)d_in[5];
  const float* beta   = (const float*)d_in[6];
  const float* rmean  = (const float*)d_in[7];
  const float* rvar   = (const float*)d_in[8];
  const float* W2     = (const float*)d_in[9];
  const float* a_src2 = (const float*)d_in[10];
  const float* a_dst2 = (const float*)d_in[11];
  const float* b2     = (const float*)d_in[12];
  const float* Wp1    = (const float*)d_in[13];
  const float* bp1    = (const float*)d_in[14];
  const float* Wp2    = (const float*)d_in[15];
  const float* bp2    = (const float*)d_in[16];
  const int*   eidx   = (const int*)d_in[17];
  const int*   edges  = (const int*)d_in[18];
  const int* src = eidx;        const int* dst = eidx + EE;
  const int* q0e = edges;       const int* q1e = edges + QQ;

  char* p = (char*)d_ws;
  float* h    = (float*)p; p += (size_t)NN * CC * 4;
  float* x1   = (float*)p; p += (size_t)NN * CC * 4;
  float* x2   = (float*)p; p += (size_t)NN * CC * 4;
  float* ssrc = (float*)p; p += (size_t)NN * 4;
  float* sdst = (float*)p; p += (size_t)NN * 4;
  unsigned* emax = (unsigned*)p; p += (size_t)NN * 4;
  float* den  = (float*)p; p += (size_t)NN * 4;
  h16* W1pk  = (h16*)p; p += (size_t)CC * CC * 2;
  h16* W2pk  = (h16*)p; p += (size_t)CC * CC * 2;
  h16* Wp1pk = (h16*)p; p += (size_t)CC * CC * 2;

  const int TB = 256;
  // weights -> f16, pre-shuffled into WMMA B-fragment order (one-time, stays in L2)
  pack_w16_kernel<<<(CC * CC + TB - 1) / TB, TB, 0, stream>>>(W1, W1pk);
  pack_w16_kernel<<<(CC * CC + TB - 1) / TB, TB, 0, stream>>>(W2, W2pk);
  pack_w16_kernel<<<(CC * CC + TB - 1) / TB, TB, 0, stream>>>(Wp1, Wp1pk);

  // ---- layer 1 ----
  gemm_node_f16<<<NN / 16, TB, 0, stream>>>(emb, W1pk, h);
  node_scores_kernel<<<(NN + 7) / 8, TB, 0, stream>>>(h, a_src1, a_dst1, ssrc, sdst);
  init_layer_kernel<<<(NN * CC + TB - 1) / TB, TB, 0, stream>>>(x1, den, emax, NN * CC);
  edge_max_kernel<<<(EE + TB - 1) / TB, TB, 0, stream>>>(src, dst, ssrc, sdst, emax);
  edge_den_kernel<<<(EE + TB - 1) / TB, TB, 0, stream>>>(src, dst, ssrc, sdst, emax, den);
  edge_agg_kernel<<<(EE + 7) / 8, TB, 0, stream>>>(src, dst, ssrc, sdst, emax, den, h, x1);
  bn_relu_kernel<<<(NN * CC + TB - 1) / TB, TB, 0, stream>>>(x1, b1, gamma, beta, rmean, rvar);

  // ---- layer 2 ----
  gemm_node_f16<<<NN / 16, TB, 0, stream>>>(x1, W2pk, h);
  node_scores_kernel<<<(NN + 7) / 8, TB, 0, stream>>>(h, a_src2, a_dst2, ssrc, sdst);
  init_layer_kernel<<<(NN * CC + TB - 1) / TB, TB, 0, stream>>>(x2, den, emax, NN * CC);
  edge_max_kernel<<<(EE + TB - 1) / TB, TB, 0, stream>>>(src, dst, ssrc, sdst, emax);
  edge_den_kernel<<<(EE + TB - 1) / TB, TB, 0, stream>>>(src, dst, ssrc, sdst, emax, den);
  edge_agg_kernel<<<(EE + 7) / 8, TB, 0, stream>>>(src, dst, ssrc, sdst, emax, den, h, x2);

  // ---- fused link predictor (b2 folded into gather) ----
  link_pred_kernel<<<QQ / 16, TB, 0, stream>>>(x2, b2, Wp1pk, bp1, Wp2, bp2,
                                               q0e, q1e, (float*)d_out);
}
